// Decoder_27848567948011
// MI455X (gfx1250) — matrix operand order repaired and checked by reading, not compiled, not run on hardware
//
#include <hip/hip_runtime.h>
#include <hip/hip_bf16.h>

typedef unsigned short u16;
typedef unsigned int   u32;
typedef __uint128_t    u128;

typedef __bf16 v16bf __attribute__((ext_vector_type(16)));
typedef float  v8f   __attribute__((ext_vector_type(8)));
typedef unsigned int u32x4 __attribute__((ext_vector_type(4)));
typedef int          i32x4 __attribute__((ext_vector_type(4)));
typedef int          i32x8 __attribute__((ext_vector_type(8)));

#define B_ 8192
#define H_ 256
#define F_ 128
#define Z_ 64
#define T_ 43
#define S_ 40

// ---------- helpers ----------

__device__ inline u16 f2bf(float f) {
  u32 u = __float_as_uint(f);
  u32 r = u + 0x7FFFu + ((u >> 16) & 1u);   // round-to-nearest-even
  return (u16)(r >> 16);
}

__device__ inline v8f zero8() {
  v8f z = {0.f, 0.f, 0.f, 0.f, 0.f, 0.f, 0.f, 0.f};
  return z;
}

union Frag { u128 q[2]; v16bf v; };

__device__ inline v8f wmma_bf(v16bf a, v16bf b, v8f c) {
  return __builtin_amdgcn_wmma_f32_16x16x32_bf16(
      false, a, false, b, (short)0, c, false, false);
}

// One 16x16x32 k-tile: A frag at pa (row-lane layout, q1 at +16 elems),
// B frag at pb (col-lane layout from W^T rows, q1 at +16 elems).
__device__ inline void ktile(v8f& acc, const u16* pa, const u16* pb) {
  Frag a, b;
  a.q[0] = *(const u128*)(pa);
  a.q[1] = *(const u128*)(pa + 16);
  b.q[0] = *(const u128*)(pb);
  b.q[1] = *(const u128*)(pb + 16);
  acc = wmma_bf(a.v, b.v, acc);
}

// One k-tile against 4 N-tiles: one shared A fragment, 4 B fragments from the
// N-tile-interleaved LDS weight slab: a lane's 4 fragment rows are 704 elems
// (1408 B) apart, so every ds_load uses a small immediate offset.
__device__ inline void ktile4(v8f& c0, v8f& c1, v8f& c2, v8f& c3,
                              const u16* pa, const u16* pw) {
  Frag a;
  a.q[0] = *(const u128*)(pa);
  a.q[1] = *(const u128*)(pa + 16);
  Frag b0, b1, b2, b3;
  b0.q[0] = *(const u128*)(pw);
  b0.q[1] = *(const u128*)(pw + 16);
  b1.q[0] = *(const u128*)(pw + 704);
  b1.q[1] = *(const u128*)(pw + 704 + 16);
  b2.q[0] = *(const u128*)(pw + 2 * 704);
  b2.q[1] = *(const u128*)(pw + 2 * 704 + 16);
  b3.q[0] = *(const u128*)(pw + 3 * 704);
  b3.q[1] = *(const u128*)(pw + 3 * 704 + 16);
  c0 = wmma_bf(a.v, b0.v, c0);
  c1 = wmma_bf(a.v, b1.v, c1);
  c2 = wmma_bf(a.v, b2.v, c2);
  c3 = wmma_bf(a.v, b3.v, c3);
}

__device__ inline float sigmoidf_(float x) { return 1.f / (1.f + expf(-x)); }

// ---------- setup kernels ----------

__global__ void zero_ws(u32* p, long n) {
  long i = (long)blockIdx.x * 256 + threadIdx.x;
  if (i < n) p[i] = 0u;
}

__global__ void cvt_bf(u16* dst, const float* src, long n) {
  long i = (long)blockIdx.x * 256 + threadIdx.x;
  if (i < n) dst[i] = f2bf(src[i]);
}

// dst[n'*dstStride + colOff + k] = bf16(src[k*N + n])  (W [K][N] -> W^T bf16).
// swz!=0: within each 64-row group, permute rows n%64 = t*16+nl -> nl*4+t so
// a lane's 4 N-tile B-fragments are adjacent in the LDS slab.
__global__ void transpose_cvt(u16* dst, int dstStride, int colOff,
                              const float* src, int K, int N, int swz) {
  long i = (long)blockIdx.x * 256 + threadIdx.x;
  if (i >= (long)K * N) return;
  int k = (int)(i / N), n = (int)(i % N);
  int nd = n;
  if (swz) nd = (n & ~63) | (((n & 15) << 2) | ((n >> 4) & 3));
  dst[(size_t)nd * dstStride + colOff + k] = f2bf(src[(size_t)k * N + n]);
}

// Hawkes intensity per (step j, batch b)
__global__ void intensity_k(const float* __restrict__ in_t,
                            const float* alpha, const float* beta, const float* mu0,
                            float* __restrict__ intens) {
  int idx = blockIdx.x * 256 + threadIdx.x;   // over S_*B_
  int j = idx >> 13;
  int b = idx & (B_ - 1);
  const float* t = in_t + (size_t)b * T_;
  float tc = t[j + 3];
  float s = 0.f;
  for (int k = 0; k < j + 3; ++k) s += expf(t[k] - tc);
  intens[idx] = mu0[0] + alpha[0] * (beta[0] * s);
}

// ---------- step kernels ----------

// gates[B,1024] = [z | h_i | y_prev | h] (K=704) @ WxWh^T.
// A 64-col x 704-K weight slab (88KB bf16, N-tile-interleaved rows) is DMA'd
// into LDS by the Tensor Data Mover once per block; each wave computes a
// 16x64 output slab: one shared A-fragment, 4 independent WMMA chains/k-tile.
__global__ __launch_bounds__(256) void gemm_lstm(
    const u16* __restrict__ zb, const u16* __restrict__ hib,
    const u16* __restrict__ ypb, const u16* __restrict__ hb,
    const u16* __restrict__ wt /*[1024][704] row-swizzled*/, float* __restrict__ gates) {
  __shared__ __align__(16) u16 ldsW[64 * 704];
  int lane = threadIdx.x;
  int ty   = threadIdx.y;
  int n0   = blockIdx.y * 64;

  if (ty == 0) {
    // Tensor DMA descriptor (D#): 2-D tensor [1024][704] bf16,
    // tile = 64 rows x 704 cols starting at row n0, dest = ldsW.
    unsigned lds_addr = (unsigned)(size_t)(void*)&ldsW[0];
    unsigned long long ga = (unsigned long long)(size_t)(wt + (size_t)n0 * 704);
    u32x4 g0;
    g0[0] = 1u;                                   // count=1 (valid user D#)
    g0[1] = lds_addr;                             // lds_addr [63:32]
    g0[2] = (u32)(ga & 0xFFFFFFFFu);              // global_addr lo
    g0[3] = (u32)((ga >> 32) & 0x01FFFFFFu) | (2u << 30);  // addr hi | type=2
    i32x8 g1;
    g1[0] = (int)(1u << 16);        // workgroup_mask=0, data_size=1 (2 bytes)
    g1[1] = (int)(704u << 16);      // atomic_addr=0 | tensor_dim0[15:0]=704
    g1[2] = (int)(1024u << 16);     // tensor_dim0 hi=0 | tensor_dim1[15:0]=1024
    g1[3] = (int)(704u << 16);      // tensor_dim1 hi=0 | tile_dim0=704
    g1[4] = 64;                     // tile_dim1=64, tile_dim2=0
    g1[5] = 704;                    // tensor_dim0_stride lo32 = 704
    g1[6] = 0;                      // stride hi | tensor_dim1_stride lo
    g1[7] = 0;
    i32x4 gz = {0, 0, 0, 0};
#if defined(__clang_major__) && (__clang_major__ >= 23)
    i32x8 gz8 = {0, 0, 0, 0, 0, 0, 0, 0};
    __builtin_amdgcn_tensor_load_to_lds(g0, g1, gz, gz, gz8, 0);
#else
    __builtin_amdgcn_tensor_load_to_lds(g0, g1, gz, gz, 0);
#endif
    __builtin_amdgcn_s_wait_tensorcnt(0);
  }
  __syncthreads();

  int m0   = (blockIdx.x * 8 + ty) * 16;
  int mrow = m0 + (lane & 15);
  int kh8  = (lane >> 4) << 3;                   // A frag K-half offset
  int kh16 = (lane >> 4) << 3;                   // B frag K-half offset
  const u16* pz = zb  + (size_t)mrow * 64  + kh8;
  const u16* ph = hib + (size_t)mrow * 256 + kh8;
  const u16* py = ypb + (size_t)mrow * 128 + kh8;
  const u16* ps = hb  + (size_t)mrow * 256 + kh8;
  // lane nl's 4 N-tile rows are at LDS rows nl*4 .. nl*4+3 (interleaved layout)
  const u16* pw = &ldsW[((lane & 15) * 4) * 704 + kh16];

  v8f c0 = zero8(), c1 = zero8(), c2 = zero8(), c3 = zero8();
#pragma unroll
  for (int t = 0; t < 2; ++t) ktile4(c0, c1, c2, c3, pz + t * 32, pw + t * 32);
#pragma unroll
  for (int t = 0; t < 8; ++t) ktile4(c0, c1, c2, c3, ph + t * 32, pw + 64 + t * 32);
#pragma unroll
  for (int t = 0; t < 4; ++t) ktile4(c0, c1, c2, c3, py + t * 32, pw + 320 + t * 32);
#pragma unroll
  for (int t = 0; t < 8; ++t) ktile4(c0, c1, c2, c3, ps + t * 32, pw + 448 + t * 32);

  int rbase = m0 + ((lane >> 4) << 3);
  int nl    = lane & 15;
#pragma unroll
  for (int i = 0; i < 8; ++i) {
    size_t rowoff = (size_t)(rbase + i) * 1024 + n0 + nl;
    gates[rowoff]      = c0[i];
    gates[rowoff + 16] = c1[i];
    gates[rowoff + 32] = c2[i];
    gates[rowoff + 48] = c3[i];
  }
}

// LSTM pointwise: c_new/h_new; carry c is intensity-gated c_new
__global__ __launch_bounds__(256) void lstm_pointwise(
    const float* __restrict__ gates, const float* __restrict__ b_lstm,
    const float* __restrict__ intens_j, float* __restrict__ cstate,
    u16* __restrict__ h_bf, u16* __restrict__ cnew_bf) {
  int idx = blockIdx.x * 256 + threadIdx.x;   // over B_*H_
  int b  = idx >> 8;
  int hc = idx & 255;
  size_t g = (size_t)b * 1024 + hc;
  float ig = gates[g]       + b_lstm[hc];
  float fg = gates[g + 256] + b_lstm[hc + 256];
  float gg = gates[g + 512] + b_lstm[hc + 512];
  float og = gates[g + 768] + b_lstm[hc + 768];
  float c_old = cstate[idx];
  float cn = sigmoidf_(fg) * c_old + sigmoidf_(ig) * tanhf(gg);
  float hn = sigmoidf_(og) * tanhf(cn);
  cstate[idx]  = intens_j[b] * cn;    // gated carry
  h_bf[idx]    = f2bf(hn);
  cnew_bf[idx] = f2bf(cn);
}

// y = relu(relu(relu(h@w1)@w2)@w3); inter-layer activations staged in LDS
__global__ __launch_bounds__(256) void ymlp(
    const u16* __restrict__ h_bf,
    const u16* __restrict__ w1t, const float* __restrict__ b1,
    const u16* __restrict__ w2t, const float* __restrict__ b2,
    const u16* __restrict__ w3t, const float* __restrict__ b3,
    float* __restrict__ ys_out, u16* __restrict__ y_bf_out, int j) {
  __shared__ __align__(16) u16 lds[2][16][128];
  int lane = threadIdx.x;
  int w    = threadIdx.y;
  int m0   = blockIdx.x * 16;
  int n0   = w * 16;
  int nl   = lane & 15;
  int col  = n0 + nl;
  int rb   = (lane >> 4) << 3;
  int kh8  = (lane >> 4) << 3;
  int kh16 = (lane >> 4) << 3;

  const u16* pa  = h_bf + (size_t)(m0 + nl) * 256 + kh8;
  const u16* pw1 = w1t + (size_t)col * 256 + kh16;
  // layer 1: K=256
  v8f acc = zero8();
#pragma unroll
  for (int t = 0; t < 8; ++t) ktile(acc, pa + t * 32, pw1 + t * 32);
  {
    float bias = b1[col];
#pragma unroll
    for (int i = 0; i < 8; ++i)
      lds[0][rb + i][col] = f2bf(fmaxf(acc[i] + bias, 0.f));
  }
  __syncthreads();
  // layer 2: K=128 from LDS
  const u16* pl0 = &lds[0][nl][kh8];
  const u16* pw2 = w2t + (size_t)col * 128 + kh16;
  acc = zero8();
#pragma unroll
  for (int t = 0; t < 4; ++t) ktile(acc, pl0 + t * 32, pw2 + t * 32);
  {
    float bias = b2[col];
#pragma unroll
    for (int i = 0; i < 8; ++i)
      lds[1][rb + i][col] = f2bf(fmaxf(acc[i] + bias, 0.f));
  }
  __syncthreads();
  // layer 3: K=128 from LDS
  const u16* pl1 = &lds[1][nl][kh8];
  const u16* pw3 = w3t + (size_t)col * 128 + kh16;
  acc = zero8();
#pragma unroll
  for (int t = 0; t < 4; ++t) ktile(acc, pl1 + t * 32, pw3 + t * 32);
  float bias = b3[col];
#pragma unroll
  for (int i = 0; i < 8; ++i) {
    float v = fmaxf(acc[i] + bias, 0.f);
    int row = m0 + rb + i;
    ys_out[((size_t)row * S_ + j) * F_ + col] = v;            // ys [B,S,F]
    y_bf_out[(size_t)row * F_ + col] = f2bf(v);
  }
}

// Fused inference (waves 0-3) + prior (waves 4-7) heads.
__global__ __launch_bounds__(256) void heads(
    const u16* __restrict__ hib, const u16* __restrict__ cnb,
    const u16* __restrict__ ycb, const u16* __restrict__ ypb,
    const u16* __restrict__ w4t, const float* __restrict__ b4,
    const u16* __restrict__ w5t, const float* __restrict__ b5,
    const u16* __restrict__ w6t, const float* __restrict__ b6,
    const u16* __restrict__ w7t, const float* __restrict__ b7,
    const u16* __restrict__ w8t, const float* __restrict__ b8,
    const u16* __restrict__ w9t, const float* __restrict__ b9,
    const float* __restrict__ eps_i, const float* __restrict__ eps_p,
    float* __restrict__ means_o, float* __restrict__ lvs_o,
    float* __restrict__ zs_o, float* __restrict__ zps_o,
    u16* __restrict__ z_bf, int j) {
  __shared__ __align__(16) u16 lds[2][16][64];
  int lane = threadIdx.x;
  int w    = threadIdx.y;
  int m0   = blockIdx.x * 16;
  int inf  = (w < 4) ? 1 : 0;
  int n0   = (w & 3) * 16;
  int nl   = lane & 15;
  int col  = n0 + nl;
  int rb   = (lane >> 4) << 3;
  int kh8  = (lane >> 4) << 3;
  int kh16 = (lane >> 4) << 3;
  int mrow = m0 + nl;

  // hidden layer of head (w4: K=768 over [h_i|c_new|y|y_prev], w7: K=640 over [h_i|c_new|y_prev])
  const u16* wt = inf ? w4t : w7t;
  int Ks = inf ? 768 : 640;
  const u16* pa_hi = hib + (size_t)mrow * 256 + kh8;
  const u16* pa_cn = cnb + (size_t)mrow * 256 + kh8;
  const u16* pa_y2 = (inf ? ycb : ypb) + (size_t)mrow * 128 + kh8;  // K 512..639
  const u16* pa_yp = ypb + (size_t)mrow * 128 + kh8;                // K 640..767 (inf only)
  const u16* pb = wt + (size_t)col * Ks + kh16;

  v8f acc = zero8();
#pragma unroll
  for (int t = 0; t < 8; ++t) ktile(acc, pa_hi + t * 32, pb + t * 32);
#pragma unroll
  for (int t = 0; t < 8; ++t) ktile(acc, pa_cn + t * 32, pb + 256 + t * 32);
#pragma unroll
  for (int t = 0; t < 4; ++t) ktile(acc, pa_y2 + t * 32, pb + 512 + t * 32);
  if (inf) {
#pragma unroll
    for (int t = 0; t < 4; ++t) ktile(acc, pa_yp + t * 32, pb + 640 + t * 32);
  }
  {
    float bias = (inf ? b4 : b7)[col];
#pragma unroll
    for (int i = 0; i < 8; ++i)
      lds[1 - inf][rb + i][col] = f2bf(fmaxf(acc[i] + bias, 0.f));
  }
  __syncthreads();

  // mean and logvar from hz (K=64)
  const u16* phz = &lds[1 - inf][nl][kh8];
  const u16* pwm = (inf ? w5t : w8t) + (size_t)col * 64 + kh16;
  const u16* pwl = (inf ? w6t : w9t) + (size_t)col * 64 + kh16;
  v8f am = zero8(), al = zero8();
#pragma unroll
  for (int t = 0; t < 2; ++t) {
    ktile(am, phz + t * 32, pwm + t * 32);
    ktile(al, phz + t * 32, pwl + t * 32);
  }
  float bm_ = (inf ? b5 : b8)[col];
  float bl_ = (inf ? b6 : b9)[col];
#pragma unroll
  for (int i = 0; i < 8; ++i) {
    int row = m0 + rb + i;
    float mean = fmaxf(am[i] + bm_, 0.f);
    float lv   = fmaxf(al[i] + bl_, 0.f);
    size_t o = ((size_t)row * S_ + j) * Z_ + col;
    if (inf) {
      float e  = eps_i[(size_t)row * Z_ + col];
      float zv = mean + e * sqrtf(expf(lv));
      means_o[o] = mean;
      lvs_o[o]   = lv;
      zs_o[o]    = zv;
      z_bf[(size_t)row * Z_ + col] = f2bf(zv);
    } else {
      float e = eps_p[(size_t)row * Z_ + col];
      zps_o[o] = mean + e * sqrtf(expf(lv));
    }
  }
}

// ---------- host ----------

extern "C" void kernel_launch(void* const* d_in, const int* in_sizes, int n_in,
                              void* d_out, int out_size, void* d_ws, size_t ws_size,
                              hipStream_t stream) {
  (void)in_sizes; (void)n_in; (void)out_size; (void)ws_size;
  const float* h_i   = (const float*)d_in[0];
  const float* in_t  = (const float*)d_in[1];
  const float* eps_i = (const float*)d_in[2];
  const float* eps_p = (const float*)d_in[3];
  const float* Wx    = (const float*)d_in[4];
  const float* Wh    = (const float*)d_in[5];
  const float* bl    = (const float*)d_in[6];
  const float* w1 = (const float*)d_in[7],  *b1 = (const float*)d_in[8];
  const float* w2 = (const float*)d_in[9],  *b2 = (const float*)d_in[10];
  const float* w3 = (const float*)d_in[11], *b3 = (const float*)d_in[12];
  const float* w4 = (const float*)d_in[13], *b4 = (const float*)d_in[14];
  const float* w5 = (const float*)d_in[15], *b5 = (const float*)d_in[16];
  const float* w6 = (const float*)d_in[17], *b6 = (const float*)d_in[18];
  const float* w7 = (const float*)d_in[19], *b7 = (const float*)d_in[20];
  const float* w8 = (const float*)d_in[21], *b8 = (const float*)d_in[22];
  const float* w9 = (const float*)d_in[23], *b9 = (const float*)d_in[24];
  const float* alpha = (const float*)d_in[25];
  const float* beta  = (const float*)d_in[26];
  const float* mu0   = (const float*)d_in[27];

  char* ws = (char*)d_ws;
  size_t off = 0;
  auto alloc = [&](size_t bytes) -> char* {
    char* p = ws + off;
    off += (bytes + 255) & ~(size_t)255;
    return p;
  };

  // zero-init region (state buffers) -- allocated first, contiguous
  float* cstate  = (float*)alloc((size_t)B_ * H_ * 4);
  u16*   h_bf    = (u16*)  alloc((size_t)B_ * H_ * 2);
  u16*   cnew_bf = (u16*)  alloc((size_t)B_ * H_ * 2);
  u16*   y0      = (u16*)  alloc((size_t)B_ * F_ * 2);
  u16*   y1      = (u16*)  alloc((size_t)B_ * F_ * 2);
  u16*   z_bf    = (u16*)  alloc((size_t)B_ * Z_ * 2);
  size_t zero_bytes = off;
  // persistent prepped data
  u16*   hi_bf  = (u16*)  alloc((size_t)B_ * H_ * 2);
  float* intens = (float*)alloc((size_t)S_ * B_ * 4);
  float* gates  = (float*)alloc((size_t)B_ * 1024 * 4);
  u16* WxWhT = (u16*)alloc((size_t)1024 * 704 * 2);
  u16* w1T = (u16*)alloc((size_t)128 * 256 * 2);
  u16* w2T = (u16*)alloc((size_t)128 * 128 * 2);
  u16* w3T = (u16*)alloc((size_t)128 * 128 * 2);
  u16* w4T = (u16*)alloc((size_t)64 * 768 * 2);
  u16* w5T = (u16*)alloc((size_t)64 * 64 * 2);
  u16* w6T = (u16*)alloc((size_t)64 * 64 * 2);
  u16* w7T = (u16*)alloc((size_t)64 * 640 * 2);
  u16* w8T = (u16*)alloc((size_t)64 * 64 * 2);
  u16* w9T = (u16*)alloc((size_t)64 * 64 * 2);

  float* out   = (float*)d_out;
  float* ys    = out;
  float* means = ys    + (size_t)B_ * S_ * F_;
  float* lvs   = means + (size_t)B_ * S_ * Z_;
  float* zs    = lvs   + (size_t)B_ * S_ * Z_;
  float* zps   = zs    + (size_t)B_ * S_ * Z_;

  // --- setup ---
  {
    long n32 = (long)(zero_bytes / 4);
    zero_ws<<<dim3((unsigned)((n32 + 255) / 256)), 256, 0, stream>>>((u32*)ws, n32);
  }
  {
    long n = (long)B_ * H_;
    cvt_bf<<<dim3((unsigned)((n + 255) / 256)), 256, 0, stream>>>(hi_bf, h_i, n);
  }
  auto tc = [&](u16* dst, int dstStride, int colOff, const float* src, int K, int N, int swz) {
    long n = (long)K * N;
    transpose_cvt<<<dim3((unsigned)((n + 255) / 256)), 256, 0, stream>>>(dst, dstStride, colOff, src, K, N, swz);
  };
  tc(WxWhT, 704, 0,   Wx, 448, 1024, 1);   // row-swizzled for LDS N-tile interleave
  tc(WxWhT, 704, 448, Wh, 256, 1024, 1);
  tc(w1T, 256, 0, w1, 256, 128, 0);
  tc(w2T, 128, 0, w2, 128, 128, 0);
  tc(w3T, 128, 0, w3, 128, 128, 0);
  tc(w4T, 768, 0, w4, 768, 64, 0);
  tc(w5T, 64,  0, w5, 64,  64, 0);
  tc(w6T, 64,  0, w6, 64,  64, 0);
  tc(w7T, 640, 0, w7, 640, 64, 0);
  tc(w8T, 64,  0, w8, 64,  64, 0);
  tc(w9T, 64,  0, w9, 64,  64, 0);
  intensity_k<<<dim3((S_ * B_) / 256), 256, 0, stream>>>(in_t, alpha, beta, mu0, intens);

  // --- sequential decode ---
  for (int j = 0; j < S_; ++j) {
    const u16* yp = (j & 1) ? y1 : y0;   // y_prev
    u16*       yc = (j & 1) ? y0 : y1;   // y (current)
    gemm_lstm<<<dim3(B_ / 128, 1024 / 64), dim3(32, 8), 0, stream>>>(
        z_bf, hi_bf, yp, h_bf, WxWhT, gates);
    lstm_pointwise<<<dim3((B_ * H_) / 256), 256, 0, stream>>>(
        gates, bl, intens + (size_t)j * B_, cstate, h_bf, cnew_bf);
    ymlp<<<dim3(B_ / 16), dim3(32, 8), 0, stream>>>(
        h_bf, w1T, b1, w2T, b2, w3T, b3, ys, yc, j);
    heads<<<dim3(B_ / 16), dim3(32, 8), 0, stream>>>(
        hi_bf, cnew_bf, yc, yp, w4T, b4, w5T, b5, w6T, b6,
        w7T, b7, w8T, b8, w9T, b9,
        eps_i + (size_t)j * B_ * Z_, eps_p + (size_t)j * B_ * Z_,
        means, lvs, zs, zps, z_bf, j);
  }
}
